// TTTLinearAdaptation_18451179503933
// MI455X (gfx1250) — matrix operand order repaired and checked by reading, not compiled, not run on hardware
//
#include <hip/hip_runtime.h>
#include <hip/hip_bf16.h>

// ---------------------------------------------------------------------------
// TTT-linear forward for MI455X (gfx1250, wave32):
//  - WMMA 16x16x32 bf16 for all matmuls
//  - async global->LDS staging (ASYNCcnt) in the GEMM
//  - Tensor Data Mover (TENSORcnt) for scan chunk staging + output store
// B=4 L=2048 DM=1024 H=16 HD=64 K=64, n_chunks=32
// ---------------------------------------------------------------------------

typedef __attribute__((ext_vector_type(16))) __bf16 bf16x16;
typedef __attribute__((ext_vector_type(8)))  float  f32x8;
typedef unsigned int u32x4 __attribute__((ext_vector_type(4)));
typedef int          i32x8 __attribute__((ext_vector_type(8)));
typedef int          i32x4 __attribute__((ext_vector_type(4)));

#define B_   4
#define L_   2048
#define DM_  1024
#define H_   16
#define HD_  64
#define KC_  64
#define NCHUNK_ 32

#if __has_builtin(__builtin_amdgcn_tensor_load_to_lds) && __has_builtin(__builtin_amdgcn_tensor_store_from_lds)
#define HAVE_TDM 1
#else
#define HAVE_TDM 0
#endif

__device__ __forceinline__ __bf16 f2bf(float x) {
  unsigned int u = __builtin_bit_cast(unsigned int, x);
  unsigned int r = u + 0x7FFFu + ((u >> 16) & 1u);   // round-to-nearest-even
  unsigned short h = (unsigned short)(r >> 16);
  return __builtin_bit_cast(__bf16, h);
}

__device__ __forceinline__ f32x8 vzero8() {
  f32x8 z = {0.f, 0.f, 0.f, 0.f, 0.f, 0.f, 0.f, 0.f};
  return z;
}

union ABFrag { bf16x16 v; unsigned int u[8]; };

// --- CDNA5 async global->LDS copy (ASYNCcnt) -------------------------------
// Generic pointers to LDS carry the LDS offset in their low 32 bits.
__device__ __forceinline__ unsigned lds_off(const void* p) {
  return (unsigned)(size_t)p;
}
__device__ __forceinline__ void async_b128(unsigned lds, const void* gptr) {
  asm volatile("global_load_async_to_lds_b128 %0, %1, off"
               :: "v"(lds), "v"(gptr) : "memory");
}
__device__ __forceinline__ void wait_async_0() {
  asm volatile("s_wait_asynccnt 0x0" ::: "memory");
}
__device__ __forceinline__ void wait_async_4() {
  asm volatile("s_wait_asynccnt 0x4" ::: "memory");
}
__device__ __forceinline__ void wait_async_8() {
  asm volatile("s_wait_asynccnt 0x8" ::: "memory");
}

#if HAVE_TDM
// --- Tensor Data Mover descriptors (ISA cdna5 ch8) -------------------------
// D# group0: [1:0]=count=1, [63:32]=lds_addr, [120:64]=global_addr, [127:126]=type=2
__device__ __forceinline__ u32x4 tdm_g0(const void* gptr, unsigned ldsoff) {
  unsigned long long ga = (unsigned long long)(size_t)gptr;
  u32x4 g0;
  g0.x = 1u;                                          // count=1 (user mode)
  g0.y = ldsoff;                                      // lds_addr (bytes)
  g0.z = (unsigned)ga;                                // global_addr lo
  g0.w = (unsigned)((ga >> 32) & 0x01FFFFFFu) | (2u << 30); // addr hi | type=2
  return g0;
}
// D# group1: workgroup_mask=0 (not in cluster), data_size, dims/tile/stride.
// width/height in data_size units; stride = tensor_dim0_stride.
__device__ __forceinline__ i32x8 tdm_g1(int dsz_log2, int width, int height, int stride) {
  i32x8 g1;
  g1[0] = dsz_log2 << 16;                             // [17:16] data_size
  g1[1] = (width & 0xFFFF) << 16;                     // tensor_dim0 lo16 @ bits 63:48
  g1[2] = ((unsigned)width >> 16) | ((height & 0xFFFF) << 16); // dim0 hi | dim1 lo
  g1[3] = ((unsigned)height >> 16) | ((width & 0xFFFF) << 16); // dim1 hi | tile_dim0
  g1[4] = height & 0xFFFF;                            // tile_dim1 (tile_dim2=0)
  g1[5] = stride;                                     // tensor_dim0_stride lo32
  g1[6] = 0;
  g1[7] = 0;
  return g1;
}
__device__ __forceinline__ void tdm_load(const void* gptr, unsigned ldsoff,
                                         int dsz_log2, int w, int h, int stride) {
  u32x4 g0 = tdm_g0(gptr, ldsoff);
  i32x8 g1 = tdm_g1(dsz_log2, w, h, stride);
  i32x4 z4 = {0, 0, 0, 0};
#if __clang_major__ >= 23
  i32x8 z8 = {0, 0, 0, 0, 0, 0, 0, 0};
  __builtin_amdgcn_tensor_load_to_lds(g0, g1, z4, z4, z8, 0);
#else
  __builtin_amdgcn_tensor_load_to_lds(g0, g1, z4, z4, 0);
#endif
}
__device__ __forceinline__ void tdm_store(void* gptr, unsigned ldsoff,
                                          int dsz_log2, int w, int h, int stride) {
  u32x4 g0 = tdm_g0(gptr, ldsoff);
  i32x8 g1 = tdm_g1(dsz_log2, w, h, stride);
  i32x4 z4 = {0, 0, 0, 0};
#if __clang_major__ >= 23
  i32x8 z8 = {0, 0, 0, 0, 0, 0, 0, 0};
  __builtin_amdgcn_tensor_store_from_lds(g0, g1, z4, z4, z8, 0);
#else
  __builtin_amdgcn_tensor_store_from_lds(g0, g1, z4, z4, 0);
#endif
}
#endif  // HAVE_TDM

// ---------------------------------------------------------------------------
// f32 -> bf16 conversion (x4 vectorized)
// ---------------------------------------------------------------------------
__global__ void cvt_f32_bf16(const float* __restrict__ in, __bf16* __restrict__ out, int n4) {
  int i = blockIdx.x * blockDim.x + threadIdx.x;
  if (i < n4) {
    float4 v = ((const float4*)in)[i];
    out[i * 4 + 0] = f2bf(v.x);
    out[i * 4 + 1] = f2bf(v.y);
    out[i * 4 + 2] = f2bf(v.z);
    out[i * 4 + 3] = f2bf(v.w);
  }
}

// ---------------------------------------------------------------------------
// bf16 GEMM: C[M,N] = A[M,K] * B[K,N], A/B bf16 row-major, C fp32.
// 128x128 tile, 8 waves, double-buffered async-to-LDS staging.
// mode 0: scatter C into [B,H,L,HD]; mode 1: plain row-major C.
// ---------------------------------------------------------------------------
__global__ __launch_bounds__(256)
void gemm_bf16_wmma(const __bf16* __restrict__ A, const __bf16* __restrict__ Bm,
                    float* __restrict__ C, int M, int N, int K, int mode)
{
  __shared__ __bf16 As[2][128][40];    // A tile row-major, padded (bank-safe)
  __shared__ __bf16 Bs[2][32][136];    // B tile row-major, padded

  const int tid  = threadIdx.x;
  const int lane = tid & 31;
  const int wave = tid >> 5;
  const int wrow = wave >> 1;
  const int wcol = wave & 1;
  const int bm = blockIdx.y * 128;
  const int bn = blockIdx.x * 128;

  f32x8 acc[2][4];
  #pragma unroll
  for (int i = 0; i < 2; ++i)
    #pragma unroll
    for (int j = 0; j < 4; ++j) acc[i][j] = vzero8();

  const int arow = tid >> 1;
  const int aseg = (tid & 1) * 16;
  const int brow = tid >> 3;
  const int bseg = (tid & 7) * 16;

  auto stage = [&](int p, int k0) {
    const __bf16* ag = A + (size_t)(bm + arow) * K + k0 + aseg;
    async_b128(lds_off(&As[p][arow][aseg]),     ag);
    async_b128(lds_off(&As[p][arow][aseg + 8]), ag + 8);
    const __bf16* bg = Bm + (size_t)(k0 + brow) * N + bn + bseg;
    async_b128(lds_off(&Bs[p][brow][bseg]),     bg);
    async_b128(lds_off(&Bs[p][brow][bseg + 8]), bg + 8);
  };

  const int am   = lane & 15;
  const int ah   = (lane >> 4) * 16;
  const int bh   = (lane >> 4) * 16;
  const int bn16 = lane & 15;

  const int nk = K / 32;
  stage(0, 0);

  for (int ik = 0; ik < nk; ++ik) {
    const int p = ik & 1;
    if (ik + 1 < nk) {
      stage(p ^ 1, (ik + 1) * 32);     // prefetch next tile (async)
      wait_async_4();                  // current tile's group complete
    } else {
      wait_async_0();
    }
    if (ik + 2 < nk)                   // far prefetch into L2
      __builtin_prefetch(A + (size_t)(bm + (tid >> 1)) * K + (ik + 2) * 32, 0, 1);
    __syncthreads();

    ABFrag af[2], bfr[4];
    #pragma unroll
    for (int i = 0; i < 2; ++i) {
      const int mrow = wrow * 32 + i * 16 + am;
      #pragma unroll
      for (int q = 0; q < 8; ++q) {
        const int offk = 2 * q + ah;
        af[i].u[q] = *(const unsigned int*)&As[p][mrow][offk];
      }
    }
    #pragma unroll
    for (int j = 0; j < 4; ++j) {
      const int ncol = wcol * 64 + j * 16 + bn16;
      #pragma unroll
      for (int q = 0; q < 8; ++q) {
        unsigned lo = *(const unsigned short*)&Bs[p][bh + 2 * q][ncol];
        unsigned hi = *(const unsigned short*)&Bs[p][bh + 2 * q + 1][ncol];
        bfr[j].u[q] = lo | (hi << 16);
      }
    }
    #pragma unroll
    for (int i = 0; i < 2; ++i)
      #pragma unroll
      for (int j = 0; j < 4; ++j)
        acc[i][j] = __builtin_amdgcn_wmma_f32_16x16x32_bf16(
            false, af[i].v, false, bfr[j].v, (short)0, acc[i][j], false, false);
    __syncthreads();
  }

  const int rbase = (lane >> 4) * 8;
  const int col   = lane & 15;
  #pragma unroll
  for (int i = 0; i < 2; ++i) {
    #pragma unroll
    for (int j = 0; j < 4; ++j) {
      #pragma unroll
      for (int r = 0; r < 8; ++r) {
        const int gm = bm + wrow * 32 + i * 16 + rbase + r;
        const int gn = bn + wcol * 64 + j * 16 + col;
        const float val = acc[i][j][r];
        if (mode == 0) {
          const int bb = gm >> 11, ll = gm & (L_ - 1);
          const int hh = gn >> 6,  dd = gn & (HD_ - 1);
          C[(((size_t)bb * H_ + hh) * L_ + ll) * HD_ + dd] = val;
        } else {
          C[(size_t)gm * N + gn] = val;
        }
      }
    }
  }
}

// ---------------------------------------------------------------------------
// 64x64x64 matmul in LDS: all 8 waves cooperate.
// ---------------------------------------------------------------------------
__device__ __forceinline__ void mm64x64x64(
    const __bf16* __restrict__ Abf, const __bf16* __restrict__ Btbf,
    int wave, int lane, f32x8& o0, f32x8& o1)
{
  const int wi = wave >> 1;
  const int jb = (wave & 1) * 2;
  o0 = vzero8(); o1 = vzero8();
  const int m  = wi * 16 + (lane & 15);
  const int ah = (lane >> 4) * 16;
  const int bh = (lane >> 4) * 16;
  const int n0 = (jb + 0) * 16 + (lane & 15);
  const int n1 = (jb + 1) * 16 + (lane & 15);
  #pragma unroll
  for (int ks = 0; ks < 2; ++ks) {
    const int k0 = ks * 32;
    ABFrag a, b0, b1;
    #pragma unroll
    for (int q = 0; q < 8; ++q) {
      const int offk = 2 * q + ah;
      a.u[q] = *(const unsigned int*)&Abf[m * 72 + k0 + offk];
    }
    #pragma unroll
    for (int q = 0; q < 8; ++q) {
      b0.u[q] = *(const unsigned int*)&Btbf[n0 * 72 + k0 + bh + 2 * q];
      b1.u[q] = *(const unsigned int*)&Btbf[n1 * 72 + k0 + bh + 2 * q];
    }
    o0 = __builtin_amdgcn_wmma_f32_16x16x32_bf16(false, a.v, false, b0.v, (short)0, o0, false, false);
    o1 = __builtin_amdgcn_wmma_f32_16x16x32_bf16(false, a.v, false, b1.v, (short)0, o1, false, false);
  }
}

// ---------------------------------------------------------------------------
// TTT scan: one block per (b,h); W/b state lives in LDS across 32 chunks.
// TDM double-buffers chunk inputs and streams chunk outputs; falls back to
// async-to-LDS loads + direct stores if the TDM builtins are unavailable.
// ---------------------------------------------------------------------------
__global__ __launch_bounds__(256)
void ttt_scan_kernel(const float* __restrict__ xq, const float* __restrict__ xk,
                     const float* __restrict__ xv,
                     const float* __restrict__ W1, const float* __restrict__ b1,
                     const float* __restrict__ theta, const float* __restrict__ theta_bias,
                     const float* __restrict__ alpha, const float* __restrict__ gma,
                     const float* __restrict__ bta,
                     __bf16* __restrict__ scan_out)
{
  extern __shared__ char smem_raw[];
  float* Wf    = (float*)smem_raw;        // [64][64] fast weights (persistent)
  float* xkf_b = Wf + 64 * 64;            // 2x [64][64] xk chunk fp32 (parity)
  float* xqf_b = xkf_b + 2 * 64 * 64;     // 2x [64][64] xq chunk fp32 (parity)
  float* Zf    = xqf_b + 2 * 64 * 64;     // [64][64] Z / grad scratch
  float* bvec  = Zf + 64 * 64;
  float* eta   = bvec + 64;
  float* ths   = eta + 64;
  float* gs    = ths + 64;
  float* bes   = gs + 64;
  __bf16* xkb  = (__bf16*)(bes + 64);     // [64][72] A: xk (later xq)
  __bf16* Wtb  = xkb + 64 * 72;           // [64][72] B: W^T
  __bf16* eTb  = Wtb + 64 * 72;           // [64][72] A: (eta*xk)^T
  __bf16* gTb  = eTb + 64 * 72;           // [64][72] B: grad^T
  __bf16* obuf = gTb + 64 * 72;           // [64][64] output staging for TDM store

  const int bhid = blockIdx.x;            // 0..63
  const int bb = bhid >> 4, hh = bhid & 15;
  const int tid = threadIdx.x, lane = tid & 31, wave = tid >> 5;

  const size_t base = (size_t)bhid * (L_ * HD_);   // into [B,H,L,HD]

  for (int i = tid; i < 4096; i += 256) Wf[i] = W1[hh * 4096 + i];
  if (tid < 64) {
    bvec[tid] = b1[hh * 64 + tid];
    ths[tid]  = theta[hh * 64 + tid];
    gs[tid]   = gma[hh * 64 + tid];
    bes[tid]  = bta[hh * 64 + tid];
  }
  const float tb = theta_bias[hh];

  // stage chunk cc's xk/xq into parity buffers
  auto stage_chunk = [&](int cc) {
    const int p = cc & 1;
    const float* kg = xk + base + (size_t)cc * (KC_ * HD_);
    const float* qg = xq + base + (size_t)cc * (KC_ * HD_);
    float* kd = xkf_b + p * 4096;
    float* qd = xqf_b + p * 4096;
#if HAVE_TDM
    // one TDM descriptor per tile, issued by one wave each (EXEC ignored)
    if (wave == 0)      tdm_load(kg, lds_off(kd), 2, 4096, 1, 4096);
    else if (wave == 1) tdm_load(qg, lds_off(qd), 2, 4096, 1, 4096);
#else
    #pragma unroll
    for (int r = 0; r < 4; ++r) {
      const int idx = (tid + r * 256) * 4;
      async_b128(lds_off(&kd[idx]), kg + idx);
      async_b128(lds_off(&qd[idx]), qg + idx);
    }
#endif
  };

  const int rbase = (lane >> 4) * 8;
  const int col   = lane & 15;
  const int wi    = wave >> 1;
  const int wjb   = (wave & 1) * 2;

  stage_chunk(0);
  __syncthreads();   // Wf / vectors ready

  for (int c = 0; c < NCHUNK_; ++c) {
    float* xkf = xkf_b + (c & 1) * 4096;
    float* xqf = xqf_b + (c & 1) * 4096;
    const float* xvg = xv + base + (size_t)c * (KC_ * HD_);

    // P0: prefetch next chunk, wait for current chunk's tiles
#if HAVE_TDM
    if (c + 1 < NCHUNK_) {
      stage_chunk(c + 1);
      if (wave < 2) __builtin_amdgcn_s_wait_tensorcnt(1);  // chunk c resident
    } else {
      if (wave < 2) __builtin_amdgcn_s_wait_tensorcnt(0);
    }
#else
    if (c + 1 < NCHUNK_) { stage_chunk(c + 1); wait_async_8(); }
    else                 { wait_async_0(); }
#endif
    __syncthreads();

    // P1: bf16 tiles (xk row-major, W transposed) + token/lr gate
    for (int i = tid; i < 4096; i += 256) {
      const int k = i >> 6, d = i & 63;
      xkb[k * 72 + d] = f2bf(xkf[i]);
      Wtb[d * 72 + k] = f2bf(Wf[i]);
    }
    if (tid < 64) {
      float dot = 0.f;
      for (int j = 0; j < 64; ++j) dot += xkf[tid * 64 + j] * ths[j];
      const float lr = 1.f / (1.f + __expf(-(dot + tb)));
      const float te = fmaxf(1.f / (float)(tid + 1) + alpha[tid], 0.f);
      eta[tid] = te * lr * (1.f / 64.f);
    }
    __syncthreads();

    // P2: Z1 = xk @ W + b
    {
      f32x8 o0, o1;
      mm64x64x64(xkb, Wtb, wave, lane, o0, o1);
      #pragma unroll
      for (int r = 0; r < 8; ++r) {
        const int mr  = wi * 16 + rbase + r;
        const int nc0 = (wjb + 0) * 16 + col;
        const int nc1 = (wjb + 1) * 16 + col;
        Zf[mr * 64 + nc0] = o0[r] + bvec[nc0];
        Zf[mr * 64 + nc1] = o1[r] + bvec[nc1];
      }
    }
    __syncthreads();

    // P3: LN + fused L2/LN backward
    if (tid < 64) {
      const int k = tid;
      float mu = 0.f, sq = 0.f;
      for (int j = 0; j < 64; ++j) { const float z = Zf[k * 64 + j]; mu += z; sq += z * z; }
      mu *= (1.f / 64.f);
      const float var  = sq * (1.f / 64.f) - mu * mu;
      const float rstd = rsqrtf(var + 1e-6f);
      float gsum = 0.f, gxsum = 0.f;
      for (int j = 0; j < 64; ++j) {
        const float xh = (Zf[k * 64 + j] - mu) * rstd;
        const float go = gs[j] * xh + bes[j] - (xvg[k * 64 + j] - xkf[k * 64 + j]);
        const float gx = go * gs[j];
        gsum += gx; gxsum += gx * xh;
      }
      const float ek = eta[k];
      for (int j = 0; j < 64; ++j) {
        const float xh = (Zf[k * 64 + j] - mu) * rstd;
        const float go = gs[j] * xh + bes[j] - (xvg[k * 64 + j] - xkf[k * 64 + j]);
        const float gx = go * gs[j];
        const float grad = (64.f * gx - gsum - xh * gxsum) * rstd * (1.f / 64.f);
        Zf[k * 64 + j] = grad;
        gTb[j * 72 + k] = f2bf(grad);
        eTb[j * 72 + k] = f2bf(ek * xkf[k * 64 + j]);
      }
    }
    __syncthreads();

    // P4: dW = (eta*xk)^T @ grad ; W -= dW ; b -= db
    {
      f32x8 o0, o1;
      mm64x64x64(eTb, gTb, wave, lane, o0, o1);
      #pragma unroll
      for (int r = 0; r < 8; ++r) {
        const int mr  = wi * 16 + rbase + r;
        const int nc0 = (wjb + 0) * 16 + col;
        const int nc1 = (wjb + 1) * 16 + col;
        Wf[mr * 64 + nc0] -= o0[r];
        Wf[mr * 64 + nc1] -= o1[r];
      }
    }
    if (tid < 64) {
      float s = 0.f;
      for (int k = 0; k < 64; ++k) s += eta[k] * Zf[k * 64 + tid];
      bvec[tid] -= s;
    }
    __syncthreads();

    // P5: refresh W^T bf16 from updated W, stage xq bf16 (reuse xkb)
    for (int i = tid; i < 4096; i += 256) {
      const int k = i >> 6, d = i & 63;
      Wtb[d * 72 + k] = f2bf(Wf[i]);
      xkb[k * 72 + d] = f2bf(xqf[i]);
    }
    __syncthreads();

    // P6: Zq = xq @ Wn + bn
    {
      f32x8 o0, o1;
      mm64x64x64(xkb, Wtb, wave, lane, o0, o1);
      #pragma unroll
      for (int r = 0; r < 8; ++r) {
        const int mr  = wi * 16 + rbase + r;
        const int nc0 = (wjb + 0) * 16 + col;
        const int nc1 = (wjb + 1) * 16 + col;
        Zf[mr * 64 + nc0] = o0[r] + bvec[nc0];
        Zf[mr * 64 + nc1] = o1[r] + bvec[nc1];
      }
    }
    __syncthreads();

#if HAVE_TDM
    // previous chunk's TDM store must be done before obuf is overwritten
    if (wave == 2) __builtin_amdgcn_s_wait_tensorcnt(0);
    __syncthreads();
#endif

    // P7: out = xq + g * LN(Zq) + be
    if (tid < 64) {
      const int k = tid;
      float mu = 0.f, sq = 0.f;
      for (int j = 0; j < 64; ++j) { const float z = Zf[k * 64 + j]; mu += z; sq += z * z; }
      mu *= (1.f / 64.f);
      const float var  = sq * (1.f / 64.f) - mu * mu;
      const float rstd = rsqrtf(var + 1e-6f);
#if HAVE_TDM
      __bf16* orow = obuf + k * 64;           // stage in LDS, TDM streams it out
#else
      __bf16* orow = scan_out + ((size_t)bb * L_ + (size_t)c * KC_ + k) * (H_ * HD_) + hh * HD_;
#endif
      for (int j = 0; j < 64; ++j) {
        const float xh = (Zf[k * 64 + j] - mu) * rstd;
        orow[j] = f2bf(xqf[k * 64 + j] + gs[j] * xh + bes[j]);
      }
    }
    __syncthreads();

#if HAVE_TDM
    // 2D TDM store: 64x64 bf16 tile, row stride 1024 elements in [B,L,H*HD]
    if (wave == 2) {
      __bf16* gout = scan_out + ((size_t)bb * L_ + (size_t)c * KC_) * (H_ * HD_) + hh * HD_;
      tdm_store(gout, lds_off(obuf), 1, 64, 64, H_ * HD_);
    }
#endif
  }

#if HAVE_TDM
  if (wave == 2) __builtin_amdgcn_s_wait_tensorcnt(0);   // drain final store
#endif
}

// ---------------------------------------------------------------------------
extern "C" void kernel_launch(void* const* d_in, const int* in_sizes, int n_in,
                              void* d_out, int out_size, void* d_ws, size_t ws_size,
                              hipStream_t stream) {
  const float* hidden     = (const float*)d_in[0];
  const float* wq         = (const float*)d_in[1];
  const float* wk         = (const float*)d_in[2];
  const float* wv         = (const float*)d_in[3];
  const float* wo         = (const float*)d_in[4];
  const float* W1         = (const float*)d_in[5];
  const float* b1         = (const float*)d_in[6];
  const float* theta      = (const float*)d_in[7];
  const float* theta_bias = (const float*)d_in[8];
  const float* alpha      = (const float*)d_in[9];
  const float* gma        = (const float*)d_in[10];
  const float* bta        = (const float*)d_in[11];

  char* ws = (char*)d_ws;
  size_t off = 0;
  auto take = [&](size_t bytes) -> void* {
    void* p = ws + off;
    off += (bytes + 255) & ~(size_t)255;
    return p;
  };
  const size_t M = (size_t)B_ * L_;  // 8192

  __bf16* hid_bf  = (__bf16*)take(M * DM_ * 2);
  __bf16* wq_bf   = (__bf16*)take((size_t)DM_ * DM_ * 2);
  __bf16* wk_bf   = (__bf16*)take((size_t)DM_ * DM_ * 2);
  __bf16* wv_bf   = (__bf16*)take((size_t)DM_ * DM_ * 2);
  __bf16* wo_bf   = (__bf16*)take((size_t)DM_ * DM_ * 2);
  float*  xq_ws   = (float*)take(M * DM_ * 4);   // [B,H,L,HD]
  float*  xk_ws   = (float*)take(M * DM_ * 4);
  float*  xv_ws   = (float*)take(M * DM_ * 4);
  __bf16* scan_bf = (__bf16*)take(M * DM_ * 2);  // [B,L,H*HD]

  const int nHid4 = (int)(M * DM_ / 4);
  const int nW4   = DM_ * DM_ / 4;
  cvt_f32_bf16<<<(nHid4 + 255) / 256, 256, 0, stream>>>(hidden, hid_bf, nHid4);
  cvt_f32_bf16<<<(nW4 + 255) / 256, 256, 0, stream>>>(wq, wq_bf, nW4);
  cvt_f32_bf16<<<(nW4 + 255) / 256, 256, 0, stream>>>(wk, wk_bf, nW4);
  cvt_f32_bf16<<<(nW4 + 255) / 256, 256, 0, stream>>>(wv, wv_bf, nW4);
  cvt_f32_bf16<<<(nW4 + 255) / 256, 256, 0, stream>>>(wo, wo_bf, nW4);

  dim3 gg(DM_ / 128, (int)(M / 128));  // (8, 64)
  gemm_bf16_wmma<<<gg, 256, 0, stream>>>(hid_bf, wq_bf, xq_ws, (int)M, DM_, DM_, 0);
  gemm_bf16_wmma<<<gg, 256, 0, stream>>>(hid_bf, wk_bf, xk_ws, (int)M, DM_, DM_, 0);
  gemm_bf16_wmma<<<gg, 256, 0, stream>>>(hid_bf, wv_bf, xv_ws, (int)M, DM_, DM_, 0);

  // smem: Wf + 2*xk + 2*xq + Zf (fp32) + 5 vec64 + 4 bf16[64][72] + obuf[64][64]
  const size_t smem = (size_t)(6 * 4096 + 5 * 64) * 4 + (size_t)(4 * 64 * 72) * 2
                    + (size_t)(64 * 64) * 2;
  ttt_scan_kernel<<<64, 256, smem, stream>>>(xq_ws, xk_ws, xv_ws, W1, b1, theta,
                                             theta_bias, alpha, gma, bta, scan_bf);

  gemm_bf16_wmma<<<gg, 256, 0, stream>>>(scan_bf, wo_bf, (float*)d_out, (int)M, DM_, DM_, 1);

  (void)in_sizes; (void)n_in; (void)out_size; (void)ws_size;
}
